// NLNN_11441792877151
// MI455X (gfx1250) — compile-verified, matrix operand order checked by reading, not run on hardware
//
#include <hip/hip_runtime.h>
#include <hip/hip_bf16.h>

typedef __attribute__((ext_vector_type(16))) __bf16 v16bf;
typedef __attribute__((ext_vector_type(8)))  __bf16 v8bf;
typedef __attribute__((ext_vector_type(8)))  float  v8f;

typedef __attribute__((ext_vector_type(4))) unsigned int tdm_g0_t;
typedef __attribute__((ext_vector_type(8))) int          tdm_g1_t;
typedef __attribute__((ext_vector_type(4))) int          tdm_g4_t;
typedef __attribute__((ext_vector_type(8))) int          tdm_g8_t;

#define NB 4
#define CC 256
#define CH 128
#define NN 4096

#define V8F_ZERO {0.f,0.f,0.f,0.f,0.f,0.f,0.f,0.f}

// Load one 16x32 bf16 WMMA fragment (A or B operand) from a row-major tile.
// Lane L (wave32): row = L&15, k-base = (L&16)?8:0; elements {kb..kb+7, kb+16..kb+23}.
__device__ __forceinline__ v16bf load_frag_rm(const __bf16* base, int stride, int lane) {
    const int r  = lane & 15;
    const int kb = (lane & 16) ? 8 : 0;
    const __bf16* p = base + r * stride + kb;
    v8bf lo = *(const v8bf*)(p);
    v8bf hi = *(const v8bf*)(p + 16);
    v16bf f;
#pragma unroll
    for (int t = 0; t < 8; ++t) { f[t] = lo[t]; f[8 + t] = hi[t]; }
    return f;
}

__device__ __forceinline__ v8f wmma_bf16(v16bf a, v16bf b, v8f c) {
    return __builtin_amdgcn_wmma_f32_16x16x32_bf16(false, a, false, b, (short)0, c, false, false);
}

// Issue a TDM 2-D tile load (global -> LDS). data_size = 2 bytes (bf16).
// D# per CDNA5 ISA 8.3/8.4: group0 {count=1, lds_addr, global_addr, type=2},
// group1 {data_size, tensor_dim0/1, tile_dim0/1, tensor_dim0_stride}.
// 6-arg builtin (clang-23 / therock): (g0, g1, g2, g3, extra, cpol).
__device__ __forceinline__ void tdm_load_2d(unsigned lds_addr, unsigned long long gaddr,
                                            unsigned tile0, unsigned tile1,
                                            unsigned tdim0, unsigned tdim1,
                                            unsigned stride0) {
    tdm_g0_t g0;
    g0[0] = 1u;                                                       // count=1 (user)
    g0[1] = lds_addr;                                                 // bits 63:32
    g0[2] = (unsigned)(gaddr & 0xffffffffull);                        // bits 95:64
    g0[3] = (unsigned)((gaddr >> 32) & 0x1ffffffull) | 0x80000000u;   // addr hi + type=2
    tdm_g1_t g1;
    g1[0] = (int)(1u << 16);                                          // data_size=1 -> 2B
    g1[1] = (int)((tdim0 & 0xffffu) << 16);                           // tensor_dim0 lo
    g1[2] = (int)(((tdim0 >> 16) & 0xffffu) | ((tdim1 & 0xffffu) << 16));
    g1[3] = (int)(((tdim1 >> 16) & 0xffffu) | ((tile0 & 0xffffu) << 16));
    g1[4] = (int)(tile1 & 0xffffu);                                   // tile_dim1, tile_dim2=0
    g1[5] = (int)stride0;                                             // dim0 stride lo32
    g1[6] = 0;
    g1[7] = 0;
    tdm_g4_t gz4 = {0, 0, 0, 0};
    tdm_g8_t gz8 = {0, 0, 0, 0, 0, 0, 0, 0};
    __builtin_amdgcn_tensor_load_to_lds(g0, g1, gz4, gz4, gz8, 0);
}

// ---- kernel 0: bf16 weight conversion + zero colsum partials -------------
__global__ void nlnn_misc(const float* __restrict__ w1, const float* __restrict__ w2,
                          const float* __restrict__ w3, const float* __restrict__ w4,
                          __bf16* __restrict__ wcatb, __bf16* __restrict__ w4b,
                          float* __restrict__ partial) {
    int idx = blockIdx.x * 256 + threadIdx.x;
    if (idx < 384 * 256) {
        int o = idx >> 8, c = idx & 255;
        float v = (o < 128) ? w1[o * 256 + c]
                : (o < 256) ? w2[(o - 128) * 256 + c]
                            : w3[(o - 256) * 256 + c];
        wcatb[idx] = (__bf16)v;
    } else if (idx < 384 * 256 + 256 * 128) {
        int i2 = idx - 384 * 256;
        w4b[i2] = (__bf16)w4[i2];
    } else if (idx < 384 * 256 + 256 * 128 + 4 * NB * NN) {
        partial[idx - (384 * 256 + 256 * 128)] = 0.f;
    }
}

// ---- kernel 1: projections  theta/phi -> [b][n][ch] bf16, g -> [b][ch][n] f32
__global__ void nlnn_prep(const float* __restrict__ x, const __bf16* __restrict__ wcatb,
                          const float* __restrict__ b1, const float* __restrict__ b2,
                          const float* __restrict__ b3,
                          __bf16* __restrict__ thetaR, __bf16* __restrict__ phiR,
                          float* __restrict__ gT0) {
    __shared__ __align__(16) __bf16 xs[32][264];   // x tile transposed to [n][c] bf16
    const int lane = threadIdx.x & 31;
    const int w    = threadIdx.x >> 5;             // 8 waves
    const int b    = blockIdx.x >> 7;              // NN/32 = 128 n-tiles per batch
    const int n0   = (blockIdx.x & 127) * 32;

    {   // cooperative transpose-stage: coalesced reads of x[c][n0..n0+31]
        int nloc = threadIdx.x & 31;
        int cgrp = threadIdx.x >> 5;
#pragma unroll 4
        for (int cc = 0; cc < 32; ++cc) {
            int c = cgrp * 32 + cc;
            xs[nloc][c] = (__bf16)x[((size_t)b * CC + c) * NN + n0 + nloc];
        }
    }
    __syncthreads();

    const int nsub  = w & 1;          // 16-n sub-tile
    const int obase = (w >> 1) * 96;  // 4 o-groups of 96 cover 384 outputs
    v8f acc[6] = {V8F_ZERO, V8F_ZERO, V8F_ZERO, V8F_ZERO, V8F_ZERO, V8F_ZERO};
#pragma unroll
    for (int k = 0; k < 8; ++k) {     // K = 256 channels, 32 per step
        v16bf a = load_frag_rm(&xs[nsub * 16][k * 32], 264, lane);
#pragma unroll
        for (int ot = 0; ot < 6; ++ot) {
            v16bf bw = load_frag_rm(wcatb + (size_t)(obase + ot * 16) * CC + k * 32, CC, lane);
            acc[ot] = wmma_bf16(a, bw, acc[ot]);
        }
    }
    const int mb = (lane >> 4) << 3;
#pragma unroll
    for (int ot = 0; ot < 6; ++ot) {
        int o = obase + ot * 16 + (lane & 15);
#pragma unroll
        for (int r = 0; r < 8; ++r) {
            int n = n0 + nsub * 16 + mb + r;
            float v = acc[ot][r];
            if (o < 128)      thetaR[((size_t)b * NN + n) * CH + o]            = (__bf16)(v + b1[o]);
            else if (o < 256) phiR  [((size_t)b * NN + n) * CH + (o - 128)]    = (__bf16)(v + b2[o - 128]);
            else              gT0   [((size_t)b * CH + (o - 256)) * NN + n]    = v + b3[o - 256];
        }
    }
}

// ---- kernel 2: colsum[j] = sum_i exp(theta_i . phi_j), 4 partials --------
__global__ void nlnn_passA(const __bf16* __restrict__ thetaR, const __bf16* __restrict__ phiR,
                           float* __restrict__ partial) {
    const int lane = threadIdx.x & 31;
    const int w    = threadIdx.x >> 5;
    const int gid  = blockIdx.x * 8 + w;     // 4096 waves
    const int b    = gid >> 10;
    const int rem  = gid & 1023;
    const int jt   = rem >> 2;               // j-tile 0..255
    const int ck   = rem & 3;                // i-chunk of 64 tiles

    v16bf pf[4];
#pragma unroll
    for (int k = 0; k < 4; ++k)
        pf[k] = load_frag_rm(phiR + ((size_t)b * NN + jt * 16) * CH + k * 32, CH, lane);

    float colacc = 0.f;
    for (int it = ck * 64; it < ck * 64 + 64; ++it) {
        v8f s = V8F_ZERO;
#pragma unroll
        for (int k = 0; k < 4; ++k) {
            v16bf ta = load_frag_rm(thetaR + ((size_t)b * NN + it * 16) * CH + k * 32, CH, lane);
            s = wmma_bf16(ta, pf[k], s);
        }
#pragma unroll
        for (int r = 0; r < 8; ++r) colacc += __expf(s[r]);
    }
    colacc += __shfl_xor(colacc, 16, 32);    // combine the two M-halves per column
    if (lane < 16)
        partial[((size_t)ck * NB + b) * NN + jt * 16 + lane] = colacc;
}

// ---- kernel 3: g' = g / colsum, bf16, [b][ch][n] -------------------------
__global__ void nlnn_scale_g(const float* __restrict__ gT0, const float* __restrict__ partial,
                             __bf16* __restrict__ gT) {
    int idx = blockIdx.x * 256 + threadIdx.x;      // b*CH*NN elements
    int j = idx & (NN - 1);
    int b = idx >> 19;                             // CH*NN = 2^19
    float s = partial[((size_t)0 * NB + b) * NN + j] + partial[((size_t)1 * NB + b) * NN + j]
            + partial[((size_t)2 * NB + b) * NN + j] + partial[((size_t)3 * NB + b) * NN + j];
    gT[idx] = (__bf16)(gT0[idx] / s);
}

// ---- kernel 4: y[n][c] = sum_j exp(theta_n . phi_j) * g'[c][j] -----------
// Staging of phi/g' tiles is done by the Tensor Data Mover, double-buffered,
// overlapped with WMMA compute; completion tracked with TENSORcnt.
__global__ void nlnn_passB(const __bf16* __restrict__ thetaR, const __bf16* __restrict__ phiR,
                           const __bf16* __restrict__ gT, __bf16* __restrict__ yR) {
    __shared__ __align__(16) __bf16 gbuf[2][CH][32];   // g' tile [c][j], double-buffered
    __shared__ __align__(16) __bf16 pbuf[2][32][CH];   // phi rows [j][ch], double-buffered
    __shared__ __align__(16) __bf16 ebuf[8][16][40];   // per-wave exp(S) re-swizzle
    const int lane = threadIdx.x & 31;
    const int w    = threadIdx.x >> 5;
    const int b    = blockIdx.x >> 5;
    const int it   = (blockIdx.x & 31) * 8 + w;        // this wave's i-tile

    v16bf taf[4];
#pragma unroll
    for (int k = 0; k < 4; ++k)
        taf[k] = load_frag_rm(thetaR + ((size_t)b * NN + it * 16) * CH + k * 32, CH, lane);

    v8f acc[8] = {V8F_ZERO, V8F_ZERO, V8F_ZERO, V8F_ZERO,
                  V8F_ZERO, V8F_ZERO, V8F_ZERO, V8F_ZERO};
    const int mb = (lane >> 4) << 3;

    // TDM issue helper: one wave issues both tile loads for j-step jp into buf.
    auto issue_tiles = [&](int jp, int buf) {
        // g' tile: 128 rows (c) x 32 elems (j), row stride NN elements
        tdm_load_2d((unsigned)(unsigned long long)&gbuf[buf][0][0],
                    (unsigned long long)(size_t)(gT + ((size_t)b * CH) * NN + jp * 32),
                    /*tile0=*/32, /*tile1=*/CH, /*tdim0=*/NN, /*tdim1=*/CH, /*stride0=*/NN);
        // phi tile: 32 consecutive rows of [CH] -> one contiguous 4096-elem run
        tdm_load_2d((unsigned)(unsigned long long)&pbuf[buf][0][0],
                    (unsigned long long)(size_t)(phiR + ((size_t)b * NN + jp * 32) * CH),
                    /*tile0=*/4096, /*tile1=*/1, /*tdim0=*/4096, /*tdim1=*/1, /*stride0=*/4096);
    };

    if (w == 0) issue_tiles(0, 0);

    for (int jp = 0; jp < 128; ++jp) {                 // 32 j per step
        const int cur = jp & 1;
        __syncthreads();                               // everyone done with buffer cur^1 (prev use)
        if (w == 0) {
            if (jp + 1 < 128) {
                issue_tiles(jp + 1, cur ^ 1);          // prefetch next pair into other buffer
                __builtin_amdgcn_s_wait_tensorcnt(2);  // in-order: current pair has landed
            } else {
                __builtin_amdgcn_s_wait_tensorcnt(0);
            }
        }
        __syncthreads();                               // release all waves onto buffer cur

#pragma unroll
        for (int jj = 0; jj < 2; ++jj) {               // two 16-j score tiles
            v8f s = V8F_ZERO;
#pragma unroll
            for (int k = 0; k < 4; ++k) {
                v16bf pb = load_frag_rm(&pbuf[cur][jj * 16][k * 32], CH, lane);
                s = wmma_bf16(taf[k], pb, s);
            }
#pragma unroll
            for (int r = 0; r < 8; ++r)                // exp + C/D->A re-swizzle via LDS
                ebuf[w][mb + r][jj * 16 + (lane & 15)] = (__bf16)__expf(s[r]);
        }
        v16bf a2 = load_frag_rm(&ebuf[w][0][0], 40, lane);
#pragma unroll
        for (int ct = 0; ct < 8; ++ct) {
            v16bf b2 = load_frag_rm(&gbuf[cur][ct * 16][0], 32, lane);
            acc[ct] = wmma_bf16(a2, b2, acc[ct]);
        }
    }
#pragma unroll
    for (int ct = 0; ct < 8; ++ct)
#pragma unroll
        for (int r = 0; r < 8; ++r) {
            int n = it * 16 + mb + r;
            int c = ct * 16 + (lane & 15);
            yR[((size_t)b * NN + n) * CH + c] = (__bf16)acc[ct][r];
        }
}

// ---- kernel 5: out = W4 * y + b4 + x -------------------------------------
__global__ void nlnn_final(const __bf16* __restrict__ yR, const __bf16* __restrict__ w4b,
                           const float* __restrict__ b4, const float* __restrict__ x,
                           float* __restrict__ out) {
    const int lane = threadIdx.x & 31;
    const int w    = threadIdx.x >> 5;     // 4 waves, 64 o each
    const int b    = blockIdx.x >> 8;
    const int n0   = (blockIdx.x & 255) * 16;

    v16bf yf[4];
#pragma unroll
    for (int k = 0; k < 4; ++k)
        yf[k] = load_frag_rm(yR + ((size_t)b * NN + n0) * CH + k * 32, CH, lane);

    v8f acc[4] = {V8F_ZERO, V8F_ZERO, V8F_ZERO, V8F_ZERO};
#pragma unroll
    for (int k = 0; k < 4; ++k)
#pragma unroll
        for (int ot = 0; ot < 4; ++ot) {
            v16bf bw = load_frag_rm(w4b + (size_t)(w * 64 + ot * 16) * CH + k * 32, CH, lane);
            acc[ot] = wmma_bf16(yf[k], bw, acc[ot]);
        }
    const int mb = (lane >> 4) << 3;
#pragma unroll
    for (int ot = 0; ot < 4; ++ot)
#pragma unroll
        for (int r = 0; r < 8; ++r) {
            int o = w * 64 + ot * 16 + (lane & 15);
            int n = n0 + mb + r;
            size_t idx = ((size_t)b * CC + o) * NN + n;
            out[idx] = acc[ot][r] + b4[o] + x[idx];
        }
}

extern "C" void kernel_launch(void* const* d_in, const int* in_sizes, int n_in,
                              void* d_out, int out_size, void* d_ws, size_t ws_size,
                              hipStream_t stream) {
    (void)in_sizes; (void)n_in; (void)out_size; (void)ws_size;
    const float* x  = (const float*)d_in[0];
    const float* w1 = (const float*)d_in[1];
    const float* b1 = (const float*)d_in[2];
    const float* w2 = (const float*)d_in[3];
    const float* b2 = (const float*)d_in[4];
    const float* w3 = (const float*)d_in[5];
    const float* b3 = (const float*)d_in[6];
    const float* w4 = (const float*)d_in[7];
    const float* b4 = (const float*)d_in[8];
    float* out = (float*)d_out;

    char* ws = (char*)d_ws;
    size_t off = 0;
    auto alloc = [&](size_t bytes) -> void* {
        void* p = ws + off;
        off = (off + bytes + 255) & ~(size_t)255;
        return p;
    };
    __bf16* wcatb   = (__bf16*)alloc((size_t)384 * 256 * 2);
    __bf16* w4b     = (__bf16*)alloc((size_t)256 * 128 * 2);
    float*  partial = (float*) alloc((size_t)4 * NB * NN * 4);
    __bf16* thetaR  = (__bf16*)alloc((size_t)NB * NN * CH * 2);
    __bf16* phiR    = (__bf16*)alloc((size_t)NB * NN * CH * 2);
    float*  gT0     = (float*) alloc((size_t)NB * CH * NN * 4);
    __bf16* gTb     = (__bf16*)alloc((size_t)NB * CH * NN * 2);
    __bf16* yR      = (__bf16*)alloc((size_t)NB * NN * CH * 2);

    nlnn_misc   <<<768,  256, 0, stream>>>(w1, w2, w3, w4, wcatb, w4b, partial);
    nlnn_prep   <<<NB * 128, 256, 0, stream>>>(x, wcatb, b1, b2, b3, thetaR, phiR, gT0);
    nlnn_passA  <<<512,  256, 0, stream>>>(thetaR, phiR, partial);
    nlnn_scale_g<<<(NB * CH * NN) / 256, 256, 0, stream>>>(gT0, partial, gTb);
    nlnn_passB  <<<NB * 32, 256, 0, stream>>>(thetaR, phiR, gTb, yR);
    nlnn_final  <<<NB * 256, 128, 0, stream>>>(yR, w4b, b4, x, out);
}
